// BuildCostVolume_14396730376755
// MI455X (gfx1250) — compile-verified, matrix operand order checked by reading, not run on hardware
//
#include <hip/hip_runtime.h>
#include <hip/hip_bf16.h>
#include <stdint.h>

#define ANG    9
#define HOUT   48
#define CIN    64
#define COUT   64
#define IMGW   432                     // ANG*HOUT
#define LSTR   72                      // padded LDS row stride (elements)
#define WELEMS (ANG * ANG * COUT * CIN)  // 331776 weight elements

typedef __attribute__((ext_vector_type(16))) __bf16       v16bf;
typedef __attribute__((ext_vector_type(8)))  float        v8f;
typedef __attribute__((ext_vector_type(4)))  unsigned int v4u;

union Frag16 { v16bf v; v4u q[2]; };

__device__ __forceinline__ unsigned short f2bf_rne(float f) {
    unsigned int u = __float_as_uint(f);
    u += 0x7FFFu + ((u >> 16) & 1u);
    return (unsigned short)(u >> 16);
}
__device__ __forceinline__ float bf2f(unsigned short h) {
    return __uint_as_float(((unsigned int)h) << 16);
}
__device__ __forceinline__ void split_bf16(float f, unsigned short &hi, unsigned short &lo) {
    hi = f2bf_rne(f);
    lo = f2bf_rne(f - bf2f(hi));
}

// disparity tap: sampled row = 9*(h + s) + a  (likewise for columns)
__device__ __forceinline__ void tap_params(int d, int t, int &s, int &a) {
    int m = d < 0 ? -d : d;
    s = m * (t - 4);
    a = (d > 0) ? (8 - t) : t;
}

// one-time pre-pass: split W[o][ci][u][v] -> hi/lo bf16 planes, layout [u*9+v][o][ci]
__global__ __launch_bounds__(256)
void split_weights(const float* __restrict__ Wt,
                   unsigned short* __restrict__ Whi,
                   unsigned short* __restrict__ Wlo) {
    const int i = blockIdx.x * 256 + threadIdx.x;
    if (i >= WELEMS) return;
    const int ci = i & 63;
    const int o  = (i >> 6) & 63;
    const int uv = i >> 12;                 // 0..80
    unsigned short hi, lo;
    split_bf16(Wt[(o * CIN + ci) * (ANG * ANG) + uv], hi, lo);
    Whi[i] = hi;
    Wlo[i] = lo;
}

template <bool PRESPLIT>
__global__ __launch_bounds__(128)
void cost_volume_wmma(const float* __restrict__ x, const float* __restrict__ Wt,
                      const unsigned short* __restrict__ Whi,
                      const unsigned short* __restrict__ Wlo,
                      float* __restrict__ out) {
    // B = gathered x panel [w][ci]; A planes only staged in LDS in fallback mode
    __shared__ __align__(16) unsigned short Ahi[PRESPLIT ? 8 : COUT * LSTR];
    __shared__ __align__(16) unsigned short Alo[PRESPLIT ? 8 : COUT * LSTR];
    __shared__ __align__(16) unsigned short Bhi[HOUT * LSTR];
    __shared__ __align__(16) unsigned short Blo[HOUT * LSTR];

    const int h  = blockIdx.x;           // output macro-row
    const int di = blockIdx.y;           // 0..8  (d = di-4)
    const int d  = di - 4;
    const int b  = blockIdx.z;

    const int tid  = threadIdx.x;
    const int wv   = tid >> 5;           // wave id 0..3 -> 16-row M tile
    const int lane = tid & 31;
    const int half = lane >> 4;          // lane group (K offset +8 in fragments)
    const int l16  = lane & 15;          // row (A) / column (B,C,D) index

    v8f acc[3] = {};                     // 3 N-tiles of 16 columns, f32 accum

    for (int u = 0; u < ANG; ++u) {
        int su, au; tap_params(d, u, su, au);
        const int hr = h + su;
        if (hr < 0 || hr >= HOUT) continue;          // uniform over block
        const int R = ANG * hr + au;                 // sampled image row

        // prefetch the full row span needed by the next u tap (27 lines/row x 64 rows)
        if (u + 1 < ANG) {
            int s2, a2; tap_params(d, u + 1, s2, a2);
            const int hr2 = h + s2;
            if (hr2 >= 0 && hr2 < HOUT) {
                const int R2 = ANG * hr2 + a2;
                for (int l = tid; l < CIN * 27; l += 128) {
                    const int ci  = l / 27;
                    const int seg = l % 27;
                    const char* p = (const char*)(x + ((size_t)(b * CIN + ci) * IMGW + R2) * IMGW)
                                    + seg * 64;
                    __builtin_prefetch(p, 0, 3);     // -> global_prefetch_b8
                }
            }
        }

        for (int v = 0; v < ANG; ++v) {
            int sv, av; tap_params(d, v, sv, av);
            __syncthreads();

            if constexpr (!PRESPLIT) {
                // fallback: stage W[:, :, u, v] -> hi/lo bf16 in LDS, layout [o][ci]
                for (int i = tid; i < COUT * CIN; i += 128) {
                    const int o  = i >> 6;
                    const int ci = i & 63;
                    const float f = Wt[(o * CIN + ci) * (ANG * ANG) + u * ANG + v];
                    unsigned short hi, lo; split_bf16(f, hi, lo);
                    Ahi[o * LSTR + ci] = hi;
                    Alo[o * LSTR + ci] = lo;
                }
            }
            // stage B: x panel -> hi/lo bf16, transposed layout [w][ci]
            for (int i = tid; i < HOUT * CIN; i += 128) {
                const int w  = i % HOUT;             // fastest over lanes -> stride-36B gather
                const int ci = i / HOUT;
                const int wc = w + sv;
                float f = 0.0f;
                if (wc >= 0 && wc < HOUT)
                    f = x[((size_t)(b * CIN + ci) * IMGW + R) * IMGW + ANG * wc + av];
                unsigned short hi, lo; split_bf16(f, hi, lo);
                Bhi[w * LSTR + ci] = hi;
                Blo[w * LSTR + ci] = lo;
            }
            __syncthreads();

            // 64 x 48 x (K=64) GEMM chunk: 2 K-steps of 32, 3 N-tiles, bf16x3
            #pragma unroll
            for (int kb = 0; kb < 2; ++kb) {
                const int cb = kb * 32 + half * 8;   // fragment K base (ISA 16-bit A layout)
                Frag16 fah, fal;
                if constexpr (PRESPLIT) {
                    // A fragments straight from global (L1-hot shared 16KB block)
                    const size_t ab = ((size_t)(u * ANG + v) * COUT + (wv * 16 + l16)) * CIN + cb;
                    fah.q[0] = *(const v4u*)&Whi[ab];
                    fah.q[1] = *(const v4u*)&Whi[ab + 16];
                    fal.q[0] = *(const v4u*)&Wlo[ab];
                    fal.q[1] = *(const v4u*)&Wlo[ab + 16];
                } else {
                    const int ar = (wv * 16 + l16) * LSTR;
                    fah.q[0] = *(const v4u*)&Ahi[ar + cb];
                    fah.q[1] = *(const v4u*)&Ahi[ar + cb + 16];
                    fal.q[0] = *(const v4u*)&Alo[ar + cb];
                    fal.q[1] = *(const v4u*)&Alo[ar + cb + 16];
                }
                #pragma unroll
                for (int nt = 0; nt < 3; ++nt) {
                    const int br = (nt * 16 + l16) * LSTR;
                    Frag16 fbh, fbl;
                    fbh.q[0] = *(const v4u*)&Bhi[br + cb];
                    fbh.q[1] = *(const v4u*)&Bhi[br + cb + 16];
                    fbl.q[0] = *(const v4u*)&Blo[br + cb];
                    fbl.q[1] = *(const v4u*)&Blo[br + cb + 16];
                    acc[nt] = __builtin_amdgcn_wmma_f32_16x16x32_bf16(
                        false, fah.v, false, fbh.v, (short)0, acc[nt], false, false);
                    acc[nt] = __builtin_amdgcn_wmma_f32_16x16x32_bf16(
                        false, fah.v, false, fbl.v, (short)0, acc[nt], false, false);
                    acc[nt] = __builtin_amdgcn_wmma_f32_16x16x32_bf16(
                        false, fal.v, false, fbh.v, (short)0, acc[nt], false, false);
                }
            }
        }
    }

    // store: out[b][o][di][h][w]; D layout: VGPR r -> M = 8*half + r, lane%16 -> N
    #pragma unroll
    for (int nt = 0; nt < 3; ++nt) {
        const int w = nt * 16 + l16;
        #pragma unroll
        for (int r = 0; r < 8; ++r) {
            const int o = wv * 16 + half * 8 + r;
            out[((((size_t)b * COUT + o) * ANG + di) * HOUT + h) * HOUT + w] = acc[nt][r];
        }
    }
}

extern "C" void kernel_launch(void* const* d_in, const int* in_sizes, int n_in,
                              void* d_out, int out_size, void* d_ws, size_t ws_size,
                              hipStream_t stream) {
    const float* x  = (const float*)d_in[0];
    const float* Wt = (const float*)d_in[1];
    float* out = (float*)d_out;
    const int B = in_sizes[0] / (CIN * IMGW * IMGW);   // 4
    dim3 grid(HOUT, ANG, B);
    dim3 block(128);

    const size_t need = (size_t)2 * WELEMS * sizeof(unsigned short);  // ~1.3 MB
    if (ws_size >= need) {
        unsigned short* Whi = (unsigned short*)d_ws;
        unsigned short* Wlo = Whi + WELEMS;
        split_weights<<<(WELEMS + 255) / 256, 256, 0, stream>>>(Wt, Whi, Wlo);
        cost_volume_wmma<true><<<grid, block, 0, stream>>>(x, Wt, Whi, Wlo, out);
    } else {
        cost_volume_wmma<false><<<grid, block, 0, stream>>>(x, Wt, nullptr, nullptr, out);
    }
}
